// fc_block_61795989455341
// MI455X (gfx1250) — matrix operand order repaired and checked by reading, not compiled, not run on hardware
//
#include <hip/hip_runtime.h>
#include <stdint.h>

typedef __bf16 v16bf __attribute__((ext_vector_type(16)));
typedef float  v8f   __attribute__((ext_vector_type(8)));

#define NROWS 8192
#define FIN   4096
#define FOUT  4096
#define BM 128
#define BN 128
#define BK 32
#define LDSS 40   // ushort stride per tile row (32 + 8 pad) = 80 bytes, 16B-aligned

#if defined(__HIP_DEVICE_COMPILE__) && \
    __has_builtin(__builtin_amdgcn_global_load_async_to_lds_b128) && \
    __has_builtin(__builtin_amdgcn_s_wait_asynccnt)
#define HAS_ASYNC 1
#else
#define HAS_ASYNC 0
#endif

#if HAS_ASYNC
// Match the builtin's parameter types exactly: GCC-vector int4 pointers in AS1/AS3.
typedef int v4i_g __attribute__((vector_size(16)));
typedef __attribute__((address_space(1))) v4i_g* gp128_t;
typedef __attribute__((address_space(3))) v4i_g* lp128_t;
#endif

union FragBF { v16bf v; uint4 q[2]; };

__device__ __forceinline__ unsigned short f2bf(float f) {
    unsigned u = __float_as_uint(f);
    unsigned r = u + 0x7FFFu + ((u >> 16) & 1u);   // round-to-nearest-even
    return (unsigned short)(r >> 16);
}
__device__ __forceinline__ float bf2f(unsigned short h) {
    return __uint_as_float(((unsigned)h) << 16);
}
__device__ __forceinline__ uint2 pack4(unsigned short a, unsigned short b,
                                       unsigned short c, unsigned short d) {
    uint2 r;
    r.x = (unsigned)a | ((unsigned)b << 16);
    r.y = (unsigned)c | ((unsigned)d << 16);
    return r;
}

// 16-byte global->LDS copy: async (ASYNCcnt-tracked, no VGPR round-trip) on CDNA5.
__device__ __forceinline__ void copy16(const unsigned short* g, unsigned short* l) {
#if HAS_ASYNC
    __builtin_amdgcn_global_load_async_to_lds_b128((gp128_t)g, (lp128_t)l, 0, 0);
#else
    *(uint4*)l = *(const uint4*)g;
#endif
}
__device__ __forceinline__ void wait_async_le6() {
#if HAS_ASYNC
    __builtin_amdgcn_s_wait_asynccnt(6);
#endif
}
__device__ __forceinline__ void wait_async_le0() {
#if HAS_ASYNC
    __builtin_amdgcn_s_wait_asynccnt(0);
#endif
}

// ---------------- kernel 0: init amax ----------------
__global__ void init_amax(unsigned* amax) {
    if (blockIdx.x == 0 && threadIdx.x == 0) *amax = 0u;
}

// ---------------- kernel 1: per-tensor abs-max of W ----------------
__global__ __launch_bounds__(256) void absmax_k(const float* __restrict__ W,
                                                unsigned* __restrict__ amax, int n) {
    __shared__ float red[256];
    float m = 0.f;
    for (int i = blockIdx.x * blockDim.x + threadIdx.x; i < n; i += gridDim.x * blockDim.x)
        m = fmaxf(m, fabsf(W[i]));
    red[threadIdx.x] = m;
    __syncthreads();
    for (int s = 128; s > 0; s >>= 1) {
        if (threadIdx.x < s) red[threadIdx.x] = fmaxf(red[threadIdx.x], red[threadIdx.x + s]);
        __syncthreads();
    }
    if (threadIdx.x == 0)
        atomicMax(amax, __float_as_uint(red[0]));   // bit order == float order for >= 0
}

// ---------------- kernel 2a: split x into bf16 hi + lo (each elem once) ----------------
__global__ __launch_bounds__(256) void prep_x(const float* __restrict__ x,
                                              unsigned short* __restrict__ xh,
                                              unsigned short* __restrict__ xl) {
    const size_t i = (size_t)blockIdx.x * blockDim.x + threadIdx.x;   // float4 index
    const float4 v = *(const float4*)(x + i * 4);
    unsigned short h0 = f2bf(v.x), h1 = f2bf(v.y), h2 = f2bf(v.z), h3 = f2bf(v.w);
    *(uint2*)(xh + i * 4) = pack4(h0, h1, h2, h3);
    *(uint2*)(xl + i * 4) = pack4(f2bf(v.x - bf2f(h0)), f2bf(v.y - bf2f(h1)),
                                  f2bf(v.z - bf2f(h2)), f2bf(v.w - bf2f(h3)));
}

// ---------------- kernel 2b: quantize W to integer-valued bf16 (each elem once) -------
__global__ __launch_bounds__(256) void prep_w(const float* __restrict__ W,
                                              unsigned short* __restrict__ wq,
                                              const int* __restrict__ wbits_p,
                                              const float* __restrict__ amax_p) {
    const int   wbits = *wbits_p;
    const float wn    = (float)((1 << (wbits - 1)) - 1);
    const float amax  = *amax_p;
    const float invs  = (amax > 0.f) ? (wn / amax) : 0.f;
    const size_t i = (size_t)blockIdx.x * blockDim.x + threadIdx.x;   // float4 index
    const float4 v = *(const float4*)(W + i * 4);
    float q0 = fminf(fmaxf(rintf(v.x * invs), -wn), wn);
    float q1 = fminf(fmaxf(rintf(v.y * invs), -wn), wn);
    float q2 = fminf(fmaxf(rintf(v.z * invs), -wn), wn);
    float q3 = fminf(fmaxf(rintf(v.w * invs), -wn), wn);
    *(uint2*)(wq + i * 4) = pack4(f2bf(q0), f2bf(q1), f2bf(q2), f2bf(q3));
}

// ---------------- kernel 3: bf16 WMMA GEMM, async double-buffered LDS ----------------
__global__ __launch_bounds__(256) void gemm_wmma(const unsigned short* __restrict__ xh,
                                                 const unsigned short* __restrict__ xl,
                                                 const unsigned short* __restrict__ wq,
                                                 const int* __restrict__ wbits_p,
                                                 const float* __restrict__ amax_p,
                                                 float* __restrict__ out) {
    __shared__ unsigned short As_hi[2][BM * LDSS];
    __shared__ unsigned short As_lo[2][BM * LDSS];
    __shared__ unsigned short Bs[2][BN * LDSS];

    const int tid  = threadIdx.x;
    const int lane = tid & 31;
    const int wid  = tid >> 5;
    const int wm   = wid >> 2;          // 0..1 -> 64 rows each
    const int wn   = wid & 3;           // 0..3 -> 32 cols each
    const int half = lane >> 4;
    const int l16  = lane & 15;
    const int bm   = blockIdx.y * BM;
    const int bn   = blockIdx.x * BN;

    const int   wbits = *wbits_p;
    const float scale = (*amax_p) / (float)((1 << (wbits - 1)) - 1);

    v8f acc[4][2];
    const v8f vz = {0.f, 0.f, 0.f, 0.f, 0.f, 0.f, 0.f, 0.f};
#pragma unroll
    for (int i = 0; i < 4; ++i)
#pragma unroll
        for (int j = 0; j < 2; ++j) acc[i][j] = vz;

    // two 16B chunks per thread per tile: 4 chunks/row * 128 rows = 512 chunks
    const int ch0  = tid;               // chunk id 0..255
    const int r0   = ch0 >> 2, c0 = (ch0 & 3) * 8;
    const int ch1  = tid + 256;         // chunk id 256..511
    const int r1   = ch1 >> 2, c1 = (ch1 & 3) * 8;

    // prologue: stage tile 0 into buffer 0
    {
        copy16(xh + (size_t)(bm + r0) * FIN + c0, &As_hi[0][r0 * LDSS + c0]);
        copy16(xh + (size_t)(bm + r1) * FIN + c1, &As_hi[0][r1 * LDSS + c1]);
        copy16(xl + (size_t)(bm + r0) * FIN + c0, &As_lo[0][r0 * LDSS + c0]);
        copy16(xl + (size_t)(bm + r1) * FIN + c1, &As_lo[0][r1 * LDSS + c1]);
        copy16(wq + (size_t)(bn + r0) * FIN + c0, &Bs[0][r0 * LDSS + c0]);
        copy16(wq + (size_t)(bn + r1) * FIN + c1, &Bs[0][r1 * LDSS + c1]);
    }

    for (int kt = 0; kt < FIN; kt += BK) {
        const int cur = (kt / BK) & 1;
        const int nxt = cur ^ 1;
        if (kt + BK < FIN) {            // stage next tile, then retire current tile's copies
            const int kk = kt + BK;
            copy16(xh + (size_t)(bm + r0) * FIN + kk + c0, &As_hi[nxt][r0 * LDSS + c0]);
            copy16(xh + (size_t)(bm + r1) * FIN + kk + c1, &As_hi[nxt][r1 * LDSS + c1]);
            copy16(xl + (size_t)(bm + r0) * FIN + kk + c0, &As_lo[nxt][r0 * LDSS + c0]);
            copy16(xl + (size_t)(bm + r1) * FIN + kk + c1, &As_lo[nxt][r1 * LDSS + c1]);
            copy16(wq + (size_t)(bn + r0) * FIN + kk + c0, &Bs[nxt][r0 * LDSS + c0]);
            copy16(wq + (size_t)(bn + r1) * FIN + kk + c1, &Bs[nxt][r1 * LDSS + c1]);
            wait_async_le6();           // in-order completion: oldest 6 (tile kt) landed
        } else {
            wait_async_le0();
        }
        __syncthreads();                // all waves' copies visible

        // ---- fragments (ISA 7.12.2 layouts) ----
        FragBF a_hi[4], a_lo[4], b[2];
#pragma unroll
        for (int mt = 0; mt < 4; ++mt) {
            const int row = wm * 64 + mt * 16 + l16;
            const uint4* ph = (const uint4*)(&As_hi[cur][row * LDSS]);
            const uint4* pl = (const uint4*)(&As_lo[cur][row * LDSS]);
            a_hi[mt].q[0] = ph[half];     a_hi[mt].q[1] = ph[2 + half];
            a_lo[mt].q[0] = pl[half];     a_lo[mt].q[1] = pl[2 + half];
        }
#pragma unroll
        for (int nt = 0; nt < 2; ++nt) {
            const int o = wn * 32 + nt * 16 + l16;
            const uint4* pb = (const uint4*)(&Bs[cur][o * LDSS]);
            b[nt].q[0] = pb[half];        b[nt].q[1] = pb[2 + half];
        }

        // ---- 16 WMMAs per k-step: hi and lo contributions ----
#pragma unroll
        for (int mt = 0; mt < 4; ++mt)
#pragma unroll
            for (int nt = 0; nt < 2; ++nt) {
                acc[mt][nt] = __builtin_amdgcn_wmma_f32_16x16x32_bf16(
                    false, a_hi[mt].v, false, b[nt].v, (short)0, acc[mt][nt], false, false);
                acc[mt][nt] = __builtin_amdgcn_wmma_f32_16x16x32_bf16(
                    false, a_lo[mt].v, false, b[nt].v, (short)0, acc[mt][nt], false, false);
            }
        __syncthreads();                // protect buf[cur] against next iteration's writes
    }

    // ---- epilogue: apply per-tensor weight scale, store fp32 ----
#pragma unroll
    for (int mt = 0; mt < 4; ++mt)
#pragma unroll
        for (int nt = 0; nt < 2; ++nt) {
            const int rbase = bm + wm * 64 + mt * 16 + half * 8;
            const int col   = bn + wn * 32 + nt * 16 + l16;
#pragma unroll
            for (int v = 0; v < 8; ++v)
                out[(size_t)(rbase + v) * FOUT + col] = acc[mt][nt][v] * scale;
        }
}

// ---------------- kernel 4: per-column batch stats ----------------
__global__ __launch_bounds__(256) void col_stats(const float* __restrict__ out,
                                                 float* __restrict__ mu,
                                                 float* __restrict__ var) {
    const int c = blockIdx.x * blockDim.x + threadIdx.x;   // coalesced down columns
    float s = 0.f, s2 = 0.f;
    for (int r = 0; r < NROWS; ++r) {
        float v = out[(size_t)r * FOUT + c];
        s += v; s2 += v * v;
    }
    const float m = s * (1.f / NROWS);
    mu[c]  = m;
    var[c] = s2 * (1.f / NROWS) - m * m;                   // biased variance
}

// ---------------- kernel 5: fused BN + clipped-ReLU + act-quant ----------------
__global__ __launch_bounds__(256) void bn_quant(float* __restrict__ out,
                                                const float* __restrict__ gamma,
                                                const float* __restrict__ beta,
                                                const float* __restrict__ mu,
                                                const float* __restrict__ var,
                                                const int* __restrict__ abits_p) {
    const int   abits = *abits_p;
    const float qn    = (float)((1 << abits) - 1);
    const float iqn   = 1.f / qn;
    const size_t i = (size_t)blockIdx.x * blockDim.x + threadIdx.x;
    const int c = (int)(i & (FOUT - 1));
    float v = out[i];
    float y = gamma[c] * (v - mu[c]) * rsqrtf(var[c] + 1e-5f) + beta[c];
    y = fminf(fmaxf(y, 0.f), 1.f);
    out[i] = rintf(y * qn) * iqn;
}

extern "C" void kernel_launch(void* const* d_in, const int* in_sizes, int n_in,
                              void* d_out, int out_size, void* d_ws, size_t ws_size,
                              hipStream_t stream) {
    const float* x     = (const float*)d_in[0];
    const float* W     = (const float*)d_in[1];
    const float* gamma = (const float*)d_in[2];
    const float* beta  = (const float*)d_in[3];
    const int*   wbits = (const int*)d_in[4];
    const int*   abits = (const int*)d_in[5];
    float* out = (float*)d_out;

    char*  wsb = (char*)d_ws;
    float* f   = (float*)wsb;
    unsigned* amax_bits = (unsigned*)wsb;                  // f[0]
    float* mu  = f + 1024;
    float* var = f + 1024 + FOUT;
    unsigned short* xh = (unsigned short*)(wsb + 65536);   // 64 MB
    unsigned short* xl = xh + (size_t)NROWS * FIN;         // 64 MB
    unsigned short* wqp = xl + (size_t)NROWS * FIN;        // 32 MB

    init_amax<<<1, 64, 0, stream>>>(amax_bits);
    absmax_k<<<1024, 256, 0, stream>>>(W, amax_bits, FOUT * FIN);
    prep_x<<<(unsigned)(((size_t)NROWS * FIN / 4) / 256), 256, 0, stream>>>(x, xh, xl);
    prep_w<<<(unsigned)(((size_t)FOUT * FIN / 4) / 256), 256, 0, stream>>>(W, wqp, wbits, f);
    gemm_wmma<<<dim3(FOUT / BN, NROWS / BM), 256, 0, stream>>>(xh, xl, wqp, wbits, f, out);
    col_stats<<<FOUT / 256, 256, 0, stream>>>(out, mu, var);
    bn_quant<<<(unsigned)(((size_t)NROWS * FOUT) / 256), 256, 0, stream>>>(
        out, gamma, beta, mu, var, abits);
}
